// RecurrentHead_15049565405872
// MI455X (gfx1250) — matrix operand order repaired and checked by reading, not compiled, not run on hardware
//
#include <hip/hip_runtime.h>
#include <hip/hip_bf16.h>
#include <stddef.h>

// ---- problem constants (from reference) ----
#define T_DIM 2048
#define B_DIM 64
#define HIN   256
#define HH    256
#define G3    768   // 3*HH (gates r,z,n)

typedef _Float16 h16_t;
typedef __attribute__((ext_vector_type(16))) _Float16 v16h;
typedef __attribute__((ext_vector_type(8)))  _Float16 v8h;
typedef __attribute__((ext_vector_type(8)))  float    v8f;

__device__ __forceinline__ float sigf(float x)   { return 1.f / (1.f + __expf(-x)); }
__device__ __forceinline__ float tanh_f(float x) { float e = __expf(2.f * x); return 1.f - 2.f / (e + 1.f); }

// ---------------------------------------------------------------------------
// K0: per-column pack. thread b walks T, records source t of each compact row.
// ---------------------------------------------------------------------------
__global__ void gru_pack(const unsigned char* __restrict__ mask,
                         int* __restrict__ src_t, int* __restrict__ lens) {
    int b = threadIdx.x;
    if (b >= B_DIM) return;
    int r = 0;
    for (int t = 0; t < T_DIM; ++t) {
        if (mask[t * B_DIM + b]) { src_t[r * B_DIM + b] = t; ++r; }
    }
    lens[b] = r;
}

// ---------------------------------------------------------------------------
// K1: exclusive prefix-sum of mask over flat (t,b) row-major order.
// ---------------------------------------------------------------------------
__global__ void gru_scan(const unsigned char* __restrict__ mask,
                         int* __restrict__ flat_rank) {
    const int N = T_DIM * B_DIM;     // 131072
    const int C = N / 1024;          // 128 per thread
    int tid  = threadIdx.x;
    int base = tid * C;
    int s = 0;
    for (int j = 0; j < C; ++j) s += mask[base + j] ? 1 : 0;
    __shared__ int sm[1024];
    sm[tid] = s;
    __syncthreads();
    for (int off = 1; off < 1024; off <<= 1) {
        int v = (tid >= off) ? sm[tid - off] : 0;
        __syncthreads();
        sm[tid] += v;
        __syncthreads();
    }
    int run = sm[tid] - s;           // exclusive prefix
    for (int j = 0; j < C; ++j) {
        int idx = base + j;
        flat_rank[idx] = run;
        run += mask[idx] ? 1 : 0;
    }
}

// ---------------------------------------------------------------------------
// K2: convert both weight matrices f32 -> f16 (row n contiguous in K = the
// WMMA B-fragment access pattern).
// ---------------------------------------------------------------------------
__global__ void gru_cvt_w(const float* __restrict__ wih, const float* __restrict__ whh,
                          h16_t* __restrict__ wih16, h16_t* __restrict__ whh16) {
    int i = blockIdx.x * blockDim.x + threadIdx.x;
    if (i < G3 * HH) {
        wih16[i] = (h16_t)wih[i];
        whh16[i] = (h16_t)whh[i];
    }
}

// ---------------------------------------------------------------------------
// K3: compact f16 activations: compact[r,b,:] = x[src_t[r,b],b,:] (0 past len)
// ---------------------------------------------------------------------------
__global__ void gru_compact(const float* __restrict__ x, const int* __restrict__ src_t,
                            const int* __restrict__ lens, h16_t* __restrict__ c16) {
    int p = blockIdx.x;              // p = r*B + b
    int c = threadIdx.x;
    int r = p >> 6, b = p & 63;
    float v = 0.f;
    if (r < lens[b]) {
        int t = src_t[p];
        v = x[(size_t)(t * B_DIM + b) * HIN + c];
    }
    c16[(size_t)p * HIN + c] = (h16_t)v;
}

// ---------------------------------------------------------------------------
// K4: gi = compact @ w_ih^T  (M=131072, N=768, K=256), V_WMMA_F32_16X16X32_F16.
// Each wave: 16x64 tile (A reused over 4 N-tiles) = 32 WMMAs.
// Result stored TRANSPOSED: gi_t[r][gcol][b]  (b contiguous) so the
// recurrence can read its 8 batch rows as one aligned v8h.
// ---------------------------------------------------------------------------
#define GI_NT 4
__global__ void __launch_bounds__(256) gru_gi(const h16_t* __restrict__ c16,
                                              const h16_t* __restrict__ wih16,
                                              h16_t* __restrict__ gi16t) {
    int wave = threadIdx.x >> 5, lane = threadIdx.x & 31;
    int task = blockIdx.x * 8 + wave;        // 98304 tasks total
    int mIdx = task / (G3 / (16 * GI_NT));   // /12
    int nG   = task % (G3 / (16 * GI_NT));
    int m0 = mIdx * 16, n0 = nG * (16 * GI_NT);
    int l15 = lane & 15;
    int akb = (lane < 16) ? 0 : 16;
    int bkb = (lane < 16) ? 0 : 16;
    const h16_t* arow = c16 + (size_t)(m0 + l15) * HIN;

    v8f acc[GI_NT] = {};
#pragma unroll
    for (int kb = 0; kb < 8; ++kb) {
        int k0 = kb * 32;
        union { v16h v; v8h h[2]; } A;
        A.h[0] = *(const v8h*)(arow + k0 + akb);
        A.h[1] = *(const v8h*)(arow + k0 + akb + 8);
#pragma unroll
        for (int t4 = 0; t4 < GI_NT; ++t4) {
            const v16h* bp = (const v16h*)(wih16 + (size_t)(n0 + t4 * 16 + l15) * HH + k0 + bkb);
            acc[t4] = __builtin_amdgcn_wmma_f32_16x16x32_f16(
                false, A.v, false, *bp, (short)0, acc[t4], false, false);
        }
    }
    int rbase = m0 + ((lane >= 16) ? 8 : 0);
#pragma unroll
    for (int t4 = 0; t4 < GI_NT; ++t4) {
        int col = n0 + t4 * 16 + l15;
#pragma unroll
        for (int v = 0; v < 8; ++v) {
            int m = rbase + v;                       // m = r*64 + b
            gi16t[((size_t)(m >> 6) * G3 + col) * B_DIM + (m & 63)] = (h16_t)acc[t4][v];
        }
    }
}

// ---------------------------------------------------------------------------
// K5: persistent GRU recurrence. 4 blocks x 256 threads; block g owns batch
// columns [16g,16g+16).  h in LDS (f32 state + f16 WMMA mirror).
//  - r/z-gate B fragments (32 x v16h = 256 VGPRs) live in registers for all
//    2048 steps.
//  - n-gate B fragments are staged ONCE into LDS (128 KB) -> per-step B
//    traffic is pure ds_read, no L2 round-trips on the serial path.
//  - gi vectors are software-pipelined one step ahead (global latency hides
//    behind the WMMA chain of the current step).
// ---------------------------------------------------------------------------
__global__ void __launch_bounds__(256) gru_rec(const float* __restrict__ hx,
                                               const float* __restrict__ bih,
                                               const float* __restrict__ bhh,
                                               const int*   __restrict__ lens,
                                               const h16_t* __restrict__ whh16,
                                               const h16_t* __restrict__ gi16t,
                                               float* __restrict__ scores,
                                               float* __restrict__ outh) {
    constexpr int LP = 264;  // padded row stride -> no LDS bank camping
    __shared__ alignas(16) float h32s[16 * LP];                 // 16.5 KB
    __shared__ alignas(16) h16_t h16s[16 * LP];                 //  8.2 KB
    __shared__ alignas(16) h16_t bNs[8 * 2 * 8 * 32 * 16];      // 128 KB

    int tid   = threadIdx.x;
    int bbase = blockIdx.x * 16;

    for (int idx = tid; idx < 16 * HH; idx += 256) {
        int i = idx >> 8, c = idx & 255;
        float v = hx[(size_t)(bbase + i) * HH + c];
        h32s[i * LP + c] = v;
        h16s[i * LP + c] = (h16_t)v;
    }

    int wave = tid >> 5, lane = tid & 31, l15 = lane & 15;
    int akb  = (lane < 16) ? 0 : 16;
    int bkb  = (lane < 16) ? 0 : 16;
    int boff = (lane >= 16) ? 8 : 0;
    int colA[2];
    colA[0] = wave * 32 + l15;
    colA[1] = colA[0] + 16;

    float biR[2], biZ[2], biN[2], bhR[2], bhZ[2], bhN[2];
#pragma unroll
    for (int t = 0; t < 2; ++t) {
        int c = colA[t];
        biR[t] = bih[c];  biZ[t] = bih[256 + c];  biN[t] = bih[512 + c];
        bhR[t] = bhh[c];  bhZ[t] = bhh[256 + c];  bhN[t] = bhh[512 + c];
    }
    int lenv[8];
#pragma unroll
    for (int v = 0; v < 8; ++v) lenv[v] = lens[bbase + boff + v];

    // ---- persistent B fragments: r and z gates, both tiles (32 x v16h) ----
    v16h bR[2][8], bZ[2][8];
#pragma unroll
    for (int t = 0; t < 2; ++t) {
        const h16_t* rRow = whh16 + (size_t)(colA[t]) * HH + bkb;
        const h16_t* zRow = whh16 + (size_t)(256 + colA[t]) * HH + bkb;
#pragma unroll
        for (int kb = 0; kb < 8; ++kb) {
            bR[t][kb] = *(const v16h*)(rRow + kb * 32);
            bZ[t][kb] = *(const v16h*)(zRow + kb * 32);
        }
    }
    // ---- stage n-gate B fragments into LDS (each lane its own 32B slots) ----
#pragma unroll
    for (int t = 0; t < 2; ++t) {
        const h16_t* nRow = whh16 + (size_t)(512 + colA[t]) * HH + bkb;
#pragma unroll
        for (int kb = 0; kb < 8; ++kb) {
            v16h tmp = *(const v16h*)(nRow + kb * 32);
            *(v16h*)(bNs + (((wave * 2 + t) * 8 + kb) * 32 + lane) * 16) = tmp;
        }
    }

    // per-lane static part of the transposed-gi offset (halfs):
    // gi_t index = r*G3*64 + gate*256*64 + colb*64 + bbase + boff
    size_t giStat[2];
    giStat[0] = (size_t)colA[0] * B_DIM + bbase + boff;
    giStat[1] = (size_t)colA[1] * B_DIM + bbase + boff;

    auto load_gi = [&](int rr, v8h (&g)[6]) {
        const h16_t* p = gi16t + (size_t)rr * (G3 * B_DIM);
        g[0] = *(const v8h*)(p + giStat[0]);
        g[1] = *(const v8h*)(p + giStat[0] + 256 * B_DIM);
        g[2] = *(const v8h*)(p + giStat[0] + 512 * B_DIM);
        g[3] = *(const v8h*)(p + giStat[1]);
        g[4] = *(const v8h*)(p + giStat[1] + 256 * B_DIM);
        g[5] = *(const v8h*)(p + giStat[1] + 512 * B_DIM);
    };

    auto step = [&](int r, v8h (&gc)[6], v8h (&gn)[6]) {
        int rn = (r + 1 < T_DIM) ? (r + 1) : r;
        load_gi(rn, gn);                 // prefetch next step's gi (hidden)

        // snapshot h A-fragments (16x256 f16) from LDS
        v16h a[8];
#pragma unroll
        for (int kb = 0; kb < 8; ++kb) {
            int kbase = kb * 32 + akb;
            union { v16h v; v8h h[2]; } A;
            A.h[0] = *(const v8h*)(h16s + l15 * LP + kbase);
            A.h[1] = *(const v8h*)(h16s + l15 * LP + kbase + 8);
            a[kb] = A.v;
        }
        __syncthreads();   // all waves captured h before anyone updates it

#pragma unroll
        for (int tile = 0; tile < 2; ++tile) {
            v8f aR = {}, aZ = {}, aN = {};
#pragma unroll
            for (int kb = 0; kb < 8; ++kb) {
                const v16h* bN = (const v16h*)(bNs + (((wave * 2 + tile) * 8 + kb) * 32 + lane) * 16);
                aN = __builtin_amdgcn_wmma_f32_16x16x32_f16(false, a[kb], false, *bN,          (short)0, aN, false, false);
                aR = __builtin_amdgcn_wmma_f32_16x16x32_f16(false, a[kb], false, bR[tile][kb], (short)0, aR, false, false);
                aZ = __builtin_amdgcn_wmma_f32_16x16x32_f16(false, a[kb], false, bZ[tile][kb], (short)0, aZ, false, false);
            }
            int colb = colA[tile];
#pragma unroll
            for (int v = 0; v < 8; ++v) {
                int i = boff + v;
                int b = bbase + i;
                float rg = sigf((float)gc[tile * 3 + 0][v] + biR[tile] + aR[v] + bhR[tile]);
                float zg = sigf((float)gc[tile * 3 + 1][v] + biZ[tile] + aZ[v] + bhZ[tile]);
                float ng = tanh_f((float)gc[tile * 3 + 2][v] + biN[tile] + rg * (aN[v] + bhN[tile]));
                float hold = h32s[i * LP + colb];
                float hn   = (1.f - zg) * ng + zg * hold;
                bool  valid = r < lenv[v];
                float hout  = valid ? hn : hold;
                h32s[i * LP + colb] = hout;
                h16s[i * LP + colb] = (h16_t)hout;
                scores[(size_t)(r * B_DIM + b) * HH + colb] = valid ? hn : 0.f;
            }
        }
        __syncthreads();   // h fully updated before next step's A loads
    };

    v8h gA[6], gB[6];
    load_gi(0, gA);
    __syncthreads();       // h + bNs staged before first step

    for (int r = 0; r < T_DIM; r += 2) {
        step(r,     gA, gB);
        step(r + 1, gB, gA);
    }

    for (int idx = tid; idx < 16 * HH; idx += 256) {
        int i = idx >> 8, c = idx & 255;
        outh[(size_t)(bbase + i) * HH + c] = h32s[i * LP + c];
    }
}

// ---------------------------------------------------------------------------
// K6: masked_scatter replication: k-th masked flat position p gets scores row k.
// ---------------------------------------------------------------------------
__global__ void gru_scatter(const unsigned char* __restrict__ mask,
                            const int* __restrict__ flat_rank,
                            const float* __restrict__ scores,
                            float* __restrict__ out) {
    int p = blockIdx.x, c = threadIdx.x;
    float v = 0.f;
    if (mask[p]) {
        int k = flat_rank[p];
        v = scores[(size_t)k * HH + c];
    }
    out[(size_t)p * HH + c] = v;
}

// ---------------------------------------------------------------------------
extern "C" void kernel_launch(void* const* d_in, const int* in_sizes, int n_in,
                              void* d_out, int out_size, void* d_ws, size_t ws_size,
                              hipStream_t stream) {
    (void)in_sizes; (void)n_in; (void)out_size; (void)ws_size;

    const float*         x    = (const float*)d_in[0];          // [T,B,256]
    const float*         hx   = (const float*)d_in[1];          // [1,B,256]
    const unsigned char* mask = (const unsigned char*)d_in[2];  // [T,B] bool (1B/elem)
    const float*         wih  = (const float*)d_in[3];          // [768,256]
    const float*         whh  = (const float*)d_in[4];          // [768,256]
    const float*         bih  = (const float*)d_in[5];          // [768]
    const float*         bhh  = (const float*)d_in[6];          // [768]

    float* out_scores = (float*)d_out;                                   // [T,B,256]
    float* out_hT     = (float*)d_out + (size_t)T_DIM * B_DIM * HH;      // [B,256]

    // workspace layout (all offsets 256B aligned)
    char* ws = (char*)d_ws;
    int*    lens    = (int*)(ws + 0);                      //       256 B
    int*    src_t   = (int*)(ws + 256);                    //   524288 B
    int*    frank   = (int*)(ws + 524544);                 //   524288 B
    h16_t*  wih16   = (h16_t*)(ws + 1048832);              //   393216 B
    h16_t*  whh16   = (h16_t*)(ws + 1442048);              //   393216 B
    h16_t*  c16     = (h16_t*)(ws + 1835264);              // 67108864 B
    h16_t*  gi16t   = (h16_t*)(ws + 68944128);             // 201326592 B
    float*  scores  = (float*)(ws + 270270720);            // 134217728 B
    // total: 404488448 B

    gru_pack   <<<1,      64,   0, stream>>>(mask, src_t, lens);
    gru_scan   <<<1,      1024, 0, stream>>>(mask, frank);
    gru_cvt_w  <<<G3 * HH / 256, 256, 0, stream>>>(wih, whh, wih16, whh16);
    gru_compact<<<T_DIM * B_DIM, 256, 0, stream>>>(x, src_t, lens, c16);
    gru_gi     <<<(T_DIM * B_DIM / 16) * (G3 / 64) / 8, 256, 0, stream>>>(c16, wih16, gi16t);
    gru_rec    <<<B_DIM / 16, 256, 0, stream>>>(hx, bih, bhh, lens, whh16, gi16t, scores, out_hT);
    gru_scatter<<<T_DIM * B_DIM, 256, 0, stream>>>(mask, frank, scores, out_scores);
}
